// ColorEmbedding_47802986004486
// MI455X (gfx1250) — matrix operand order repaired and checked by reading, not compiled, not run on hardware
//
#include <hip/hip_runtime.h>
#include <hip/hip_fp16.h>

typedef __attribute__((ext_vector_type(16))) _Float16 v16h;
typedef __attribute__((ext_vector_type(8)))  _Float16 v8h;
typedef __attribute__((ext_vector_type(8)))  float    v8f;

// ---------------------------------------------------------------------------
// Weight transform: OIHW f32 -> [tap = r*4+s][Cout][Cin-swizzled] f16.
// Within each 32-wide K chunk, channels are stored in WMMA A-fragment order:
//   positions [0..15]  (lane-half 0): ci = {0..15}
//   positions [16..31] (lane-half 1): ci = {16..31}
// so each lane's 16 A values are one contiguous 32-byte load.
// ---------------------------------------------------------------------------
__global__ void wtrans_kernel(const float* __restrict__ w, _Float16* __restrict__ wt,
                              int Cout, int Cin) {
    size_t idx = (size_t)blockIdx.x * blockDim.x + threadIdx.x;
    size_t total = (size_t)Cout * Cin * 16;
    if (idx >= total) return;
    int cpos = (int)(idx % Cin);            // swizzled channel position
    int o    = (int)((idx / Cin) % Cout);
    int t    = (int)(idx / ((size_t)Cin * Cout));
    int r = t >> 2, s = t & 3;
    // Unswizzle: position -> original input channel
    int chunk = cpos >> 5;
    int within = cpos & 31;
    int lh = within >> 4;
    int e  = within & 15;
    int ci_in_chunk = 16 * lh + e;
    int ci = chunk * 32 + ci_in_chunk;
    wt[idx] = (_Float16)w[(((size_t)o * Cin + ci) * 4 + r) * 4 + s];
}

// ---------------------------------------------------------------------------
// Conv1: direct (Cin=3, 3% of FLOPs). img NCHW f32 -> act1 NHWC f16 (ReLU).
// [64,3,256,256] -> [64,128,128,64]
// ---------------------------------------------------------------------------
__global__ void conv1_kernel(const float* __restrict__ img, const float* __restrict__ w1,
                             const float* __restrict__ b1, _Float16* __restrict__ act1) {
    size_t idx = (size_t)blockIdx.x * blockDim.x + threadIdx.x;
    if (idx >= (size_t)64 * 128 * 128 * 64) return;
    const int co = (int)(idx & 63);
    size_t p = idx >> 6;
    const int wo = (int)(p & 127);
    const int ho = (int)((p >> 7) & 127);
    const int b  = (int)(p >> 14);
    float sum = b1[co];
    for (int ci = 0; ci < 3; ++ci) {
#pragma unroll
        for (int r = 0; r < 4; ++r) {
            const int hi = 2 * ho - 1 + r;
            if ((unsigned)hi >= 256u) continue;
#pragma unroll
            for (int s = 0; s < 4; ++s) {
                const int wi = 2 * wo - 1 + s;
                if ((unsigned)wi >= 256u) continue;
                sum += img[(((size_t)b * 3 + ci) * 256 + hi) * 256 + wi] *
                       w1[((co * 3 + ci) * 4 + r) * 4 + s];
            }
        }
    }
    sum = sum > 0.f ? sum : 0.f;
    act1[(((size_t)b * 128 + ho) * 128 + wo) * 64 + co] = (_Float16)sum;
}

// ---------------------------------------------------------------------------
// Implicit-GEMM conv via V_WMMA_F32_16X16X32_F16 with 2x4 register blocking.
// Each wave: 32 output channels x 64 output pixels.
//   M = out channels (two 16-wide tiles), N = flattened out pixels (four
//   16-wide tiles), K = Cin in chunks of 32; 16 taps accumulate via WMMAs.
// act NHWC f16, wt [tap][Cout][Cin-swizzled] f16. One wave per block.
// A fragment: one contiguous v16h load (weights pre-swizzled to lane order).
// B fragment: one contiguous v16h load, N = lane&15, elems = K 16h+0..15.
// D: lane l, VGPR v -> channel = v + 8*(l>>4), pixel = l&15.
// ---------------------------------------------------------------------------
__global__ __launch_bounds__(32) void conv_wmma_kernel(
    const _Float16* __restrict__ act_in, const _Float16* __restrict__ wt,
    const float* __restrict__ bias, _Float16* __restrict__ act_out,
    int Cin, int Cout, int Hin, int Win, int Hout, int Wout) {
    const int lane = threadIdx.x;
    const int lh   = lane >> 4;        // lane half: K-group selector
    const int l16  = lane & 15;
    const int b    = blockIdx.y;
    const int co0  = blockIdx.z << 5;  // 32 channels per block
    const int p0   = blockIdx.x << 6;  // 64 flattened pixels per block

    // Per-lane pixel coords for the four N tiles
    int hoj[4], woj[4];
#pragma unroll
    for (int j = 0; j < 4; ++j) {
        const int p = p0 + j * 16 + l16;
        hoj[j] = p / Wout;
        woj[j] = p % Wout;
    }

    v8f acc[2][4] = {{{}, {}, {}, {}}, {{}, {}, {}, {}}};
    const size_t tapStride = (size_t)Cout * Cin;

    for (int kc = 0; kc < Cin; kc += 32) {
        // Prefetch next K-chunk of weights into cache (global_prefetch_b8)
        if (kc + 32 < Cin) {
            __builtin_prefetch(
                (const void*)(wt + (size_t)(co0 + l16) * Cin + kc + 32), 0, 3);
        }
#pragma unroll
        for (int t = 0; t < 16; ++t) {
            const int r = t >> 2, s = t & 3;
            // A fragments: two co tiles, one contiguous v16h load each
            v16h a[2];
#pragma unroll
            for (int i = 0; i < 2; ++i) {
                a[i] = *(const v16h*)(wt + t * tapStride +
                       (size_t)(co0 + 16 * i + l16) * Cin + kc + 16 * lh);
            }
            // B fragments: four pixel tiles, one 32B load each (zero at padding)
            v16h bf[4];
#pragma unroll
            for (int j = 0; j < 4; ++j) {
                const int hi_ = 2 * hoj[j] - 1 + r;
                const int wi_ = 2 * woj[j] - 1 + s;
                v16h v = {};
                if ((unsigned)hi_ < (unsigned)Hin && (unsigned)wi_ < (unsigned)Win) {
                    v = *(const v16h*)(act_in +
                        (((size_t)b * Hin + hi_) * Win + wi_) * Cin + kc + 16 * lh);
                }
                bf[j] = v;
            }
            // 8 WMMAs: A reused 4x, B reused 2x
#pragma unroll
            for (int i = 0; i < 2; ++i)
#pragma unroll
                for (int j = 0; j < 4; ++j)
                    acc[i][j] = __builtin_amdgcn_wmma_f32_16x16x32_f16(
                        false, a[i], false, bf[j], (short)0, acc[i][j], false, false);
        }
    }

    // Store: 8 contiguous channels per lane per (i,j), bias + ReLU + cvt f16
#pragma unroll
    for (int i = 0; i < 2; ++i) {
#pragma unroll
        for (int j = 0; j < 4; ++j) {
            v8h outv;
#pragma unroll
            for (int v = 0; v < 8; ++v) {
                float x = acc[i][j][v] + bias[co0 + 16 * i + 8 * lh + v];
                x = x > 0.f ? x : 0.f;
                outv[v] = (_Float16)x;
            }
            *(v8h*)(act_out + (((size_t)b * Hout + hoj[j]) * Wout + woj[j]) * Cout +
                    co0 + 16 * i + 8 * lh) = outv;
        }
    }
}

// ---------------------------------------------------------------------------
// Histogram: one block per (b,c); 16 bins over [-1,1] of (img+1)*0.5; L1-norm.
// ---------------------------------------------------------------------------
__global__ void hist_kernel(const float* __restrict__ img, float* __restrict__ hist) {
    __shared__ unsigned cnt[16];
    __shared__ float norm;
    const int bc = blockIdx.x;       // b*3 + c
    const int tid = threadIdx.x;     // 256 threads
    if (tid < 16) cnt[tid] = 0u;
    __syncthreads();
    const float* p = img + (size_t)bc * 65536;
    for (int i = tid; i < 65536; i += 256) {
        float x = (p[i] + 1.0f) * 0.5f;
        int bin = (int)floorf((x + 1.0f) * 8.0f);   // width = 2/16
        bin = bin < 0 ? 0 : (bin > 15 ? 15 : bin);
        if (x >= -1.0f && x <= 1.0f) atomicAdd(&cnt[bin], 1u);
    }
    __syncthreads();
    if (tid == 0) {
        unsigned tot = 0;
        for (int i = 0; i < 16; ++i) tot += cnt[i];
        norm = 1.0f / ((float)tot + 1e-6f);
    }
    __syncthreads();
    if (tid < 16) hist[(size_t)bc * 16 + tid] = (float)cnt[tid] * norm;
}

// ---------------------------------------------------------------------------
// Global average pool: act4 NHWC f16 [64,16,16,512] -> pooled f32 [64,512]
// ---------------------------------------------------------------------------
__global__ void pool_kernel(const _Float16* __restrict__ act4, float* __restrict__ pooled) {
    const int b = blockIdx.x;
    for (int ch = threadIdx.x; ch < 512; ch += blockDim.x) {
        float sum = 0.f;
        const _Float16* p = act4 + (size_t)b * 256 * 512 + ch;
        for (int px = 0; px < 256; ++px) sum += (float)p[(size_t)px * 512];
        pooled[b * 512 + ch] = sum * (1.0f / 256.0f);
    }
}

// ---------------------------------------------------------------------------
// Head: img_feat = pooled@wf.T+bf; hist_feat = hist@wh.T+bh; color = comb@wo.T+bo
// ---------------------------------------------------------------------------
__global__ __launch_bounds__(64) void head_kernel(
    const float* __restrict__ pooled, const float* __restrict__ hist,
    const float* __restrict__ wf, const float* __restrict__ bf,
    const float* __restrict__ wh, const float* __restrict__ bh,
    const float* __restrict__ w_o, const float* __restrict__ b_o,
    float* __restrict__ color) {
    __shared__ float comb[128];
    const int b = blockIdx.x;
    const int tid = threadIdx.x;  // 64
    float s1 = bf[tid];
    for (int k = 0; k < 512; ++k) s1 += pooled[b * 512 + k] * wf[tid * 512 + k];
    comb[tid] = s1;
    float s2 = bh[tid];
    for (int k = 0; k < 48; ++k) s2 += hist[b * 48 + k] * wh[tid * 48 + k];
    comb[64 + tid] = s2;
    __syncthreads();
    if (tid < 3) {
        float s = b_o[tid];
        for (int k = 0; k < 128; ++k) s += comb[k] * w_o[tid * 128 + k];
        color[b * 3 + tid] = s;
    }
}

// ---------------------------------------------------------------------------
// Broadcast color_feat [64,3] -> color_map [64,3,256,256]
// ---------------------------------------------------------------------------
__global__ void bcast_kernel(const float* __restrict__ color, float* __restrict__ out) {
    size_t idx = (size_t)blockIdx.x * blockDim.x + threadIdx.x;
    if (idx >= (size_t)64 * 3 * 65536) return;
    out[idx] = color[idx >> 16];
}

// ---------------------------------------------------------------------------
extern "C" void kernel_launch(void* const* d_in, const int* in_sizes, int n_in,
                              void* d_out, int out_size, void* d_ws, size_t ws_size,
                              hipStream_t stream) {
    (void)in_sizes; (void)n_in; (void)out_size; (void)ws_size;
    const float* img = (const float*)d_in[0];
    const float* w1  = (const float*)d_in[1];
    const float* b1  = (const float*)d_in[2];
    const float* w2  = (const float*)d_in[3];
    const float* b2  = (const float*)d_in[4];
    const float* w3  = (const float*)d_in[5];
    const float* b3  = (const float*)d_in[6];
    const float* w4  = (const float*)d_in[7];
    const float* b4  = (const float*)d_in[8];
    const float* wf  = (const float*)d_in[9];
    const float* bf  = (const float*)d_in[10];
    const float* wh  = (const float*)d_in[11];
    const float* bh  = (const float*)d_in[12];
    const float* w_o = (const float*)d_in[13];
    const float* b_o = (const float*)d_in[14];
    float* out = (float*)d_out;

    char* ws = (char*)d_ws;
    size_t off = 0;
    auto alloc = [&](size_t bytes) -> char* {
        char* p = ws + off;
        off = (off + bytes + 255) & ~(size_t)255;
        return p;
    };
    _Float16* wt2  = (_Float16*)alloc((size_t)128 * 64 * 16 * 2);
    _Float16* wt3  = (_Float16*)alloc((size_t)256 * 128 * 16 * 2);
    _Float16* wt4  = (_Float16*)alloc((size_t)512 * 256 * 16 * 2);
    _Float16* bufA = (_Float16*)alloc((size_t)64 * 128 * 128 * 64 * 2); // act1 / act3
    _Float16* bufB = (_Float16*)alloc((size_t)64 * 64 * 64 * 128 * 2);  // act2 / act4
    float* hist    = (float*)alloc((size_t)64 * 48 * 4);
    float* pooled  = (float*)alloc((size_t)64 * 512 * 4);
    float* color   = (float*)alloc((size_t)64 * 3 * 4);
    _Float16* act1 = bufA, *act3 = bufA;
    _Float16* act2 = bufB, *act4 = bufB;

    // Weight transforms (fragment-order swizzle)
    wtrans_kernel<<<(128 * 64 * 16 + 255) / 256, 256, 0, stream>>>(w2, wt2, 128, 64);
    wtrans_kernel<<<(256 * 128 * 16 + 255) / 256, 256, 0, stream>>>(w3, wt3, 256, 128);
    wtrans_kernel<<<(512 * 256 * 16 + 255) / 256, 256, 0, stream>>>(w4, wt4, 512, 256);

    // Conv1 (direct)
    size_t n1 = (size_t)64 * 128 * 128 * 64;
    conv1_kernel<<<(unsigned)((n1 + 255) / 256), 256, 0, stream>>>(img, w1, b1, act1);

    // Conv2/3/4 (WMMA implicit GEMM, 2x4 blocked):
    //   grid (P/64, B, Cout/32), 1 wave per block
    conv_wmma_kernel<<<dim3((64 * 64) / 64, 64, 128 / 32), 32, 0, stream>>>(
        act1, wt2, b2, act2, 64, 128, 128, 128, 64, 64);
    conv_wmma_kernel<<<dim3((32 * 32) / 64, 64, 256 / 32), 32, 0, stream>>>(
        act2, wt3, b3, act3, 128, 256, 64, 64, 32, 32);
    conv_wmma_kernel<<<dim3((16 * 16) / 64, 64, 512 / 32), 32, 0, stream>>>(
        act3, wt4, b4, act4, 256, 512, 32, 32, 16, 16);

    // Head
    hist_kernel<<<192, 256, 0, stream>>>(img, hist);
    pool_kernel<<<64, 256, 0, stream>>>(act4, pooled);
    head_kernel<<<64, 64, 0, stream>>>(pooled, hist, wf, bf, wh, bh, w_o, b_o, color);
    size_t no = (size_t)64 * 3 * 65536;
    bcast_kernel<<<(unsigned)((no + 255) / 256), 256, 0, stream>>>(color, out);
}
